// Focal_Transformer_53283364274508
// MI455X (gfx1250) — matrix operand order repaired and checked by reading, not compiled, not run on hardware
//
#include <hip/hip_runtime.h>
#include <hip/hip_bf16.h>

#define BATCH   8
#define CC      256
#define NN      4096
#define INDIM   512
#define HEADS   8
#define HDIM    64
#define N2      1024
#define N3      256
#define NKVTOT  5120
#define LNEPS   1e-5f
#define USE_ASYNC_LDS 1

typedef __attribute__((ext_vector_type(16))) __bf16 v16bf;
typedef __attribute__((ext_vector_type(8)))  __bf16 v8bf;
typedef __attribute__((ext_vector_type(8)))  float  v8f;

// ---------------------------------------------------------------- fragments
__device__ __forceinline__ v16bf frag_ab(const __bf16* p0, const __bf16* p1) {
  v8bf a = *(const v8bf*)p0;
  v8bf b = *(const v8bf*)p1;
  v16bf r;
#pragma unroll
  for (int i = 0; i < 8; ++i) { r[i] = a[i]; r[i + 8] = b[i]; }
  return r;
}
__device__ __forceinline__ v16bf frag_c16(const __bf16* p) { return frag_ab(p, p + 8); }

// A: row-major [M][K] (Arow already offset to this lane's row = lane&15)
// B: N-major  [N][K]  (Bcol already offset to col = tileN + (lane&15))
// Manual 2x-unrolled ping-pong: loads for the next half-step land directly in
// the fragment registers the following WMMAs consume -> no rotation moves, and
// waits stay partial (loads of one set overlap WMMAs of the other).
template <int K>
__device__ __forceinline__ void wmma_kloop(const __bf16* __restrict__ Arow,
                                           const __bf16* __restrict__ Bcol,
                                           int hlf, v8f acc[4]) {
  static_assert(K % 64 == 0, "K must be a multiple of 64");
  v16bf afA = frag_ab(Arow + 16 * hlf, Arow + 8 + 16 * hlf);
  v16bf bfA[4];
#pragma unroll
  for (int nt = 0; nt < 4; ++nt)
    bfA[nt] = frag_c16(Bcol + (size_t)nt * 16 * K + 16 * hlf);
  for (int k = 0; k < K; k += 64) {
    v16bf afB = frag_ab(Arow + k + 32 + 16 * hlf, Arow + k + 40 + 16 * hlf);
    v16bf bfB[4];
#pragma unroll
    for (int nt = 0; nt < 4; ++nt)
      bfB[nt] = frag_c16(Bcol + (size_t)nt * 16 * K + k + 32 + 16 * hlf);
#pragma unroll
    for (int nt = 0; nt < 4; ++nt)
      acc[nt] = __builtin_amdgcn_wmma_f32_16x16x32_bf16(
          false, afA, false, bfA[nt], (short)0, acc[nt], false, false);
    if (k + 64 < K) {
      afA = frag_ab(Arow + k + 64 + 16 * hlf, Arow + k + 72 + 16 * hlf);
#pragma unroll
      for (int nt = 0; nt < 4; ++nt)
        bfA[nt] = frag_c16(Bcol + (size_t)nt * 16 * K + k + 64 + 16 * hlf);
    }
#pragma unroll
    for (int nt = 0; nt < 4; ++nt)
      acc[nt] = __builtin_amdgcn_wmma_f32_16x16x32_bf16(
          false, afB, false, bfB[nt], (short)0, acc[nt], false, false);
  }
}

__device__ __forceinline__ void zero4(v8f acc[4]) {
#pragma unroll
  for (int nt = 0; nt < 4; ++nt)
#pragma unroll
    for (int i = 0; i < 8; ++i) acc[nt][i] = 0.f;
}

// ------------------------------------------------- async global->LDS (CDNA5)
#if USE_ASYNC_LDS
__device__ __forceinline__ void async_ld_b128(unsigned lds_off, unsigned goff,
                                              const void* sbase) {
  // GVS form: LDS-offset VGPR, 32-bit global-offset VGPR, 64-bit SGPR base
  asm volatile("global_load_async_to_lds_b128 %0, %1, %2 offset:0"
               :: "v"(lds_off), "v"(goff), "s"(sbase) : "memory");
}
__device__ __forceinline__ void wait_async0() {
  asm volatile("s_wait_asynccnt 0x0" ::: "memory");
}
#endif

// ---------------------------------------------------------------- weight cvt
// w [K,N] f32 -> wt [N,K] bf16
__global__ void __launch_bounds__(256) wcvt_kernel(const float* __restrict__ w,
                                                   __bf16* __restrict__ wt,
                                                   int K, int N) {
  int idx = blockIdx.x * 256 + threadIdx.x;
  if (idx >= K * N) return;
  int n = idx % N, k = idx / N;
  wt[(size_t)n * K + k] = (__bf16)w[idx];
}

// ---------------------------------------------------------------- LN1 (C=256)
__global__ void __launch_bounds__(256) ln1_kernel(const float* __restrict__ x,
                                                  const float* __restrict__ g,
                                                  const float* __restrict__ bb,
                                                  __bf16* __restrict__ kv) {
  int lane = threadIdx.x & 31, wv = threadIdx.x >> 5;
  long row = (long)blockIdx.x * 8 + wv;            // 0 .. B*NN-1
  int b = (int)(row >> 12), m = (int)(row & (NN - 1));
  const float* xr = x + (size_t)row * CC;
  float v[8], s = 0.f;
#pragma unroll
  for (int i = 0; i < 8; ++i) { v[i] = xr[lane * 8 + i]; s += v[i]; }
#pragma unroll
  for (int off = 16; off; off >>= 1) s += __shfl_xor(s, off, 32);
  float mean = s * (1.f / CC), sq = 0.f;
#pragma unroll
  for (int i = 0; i < 8; ++i) { float d = v[i] - mean; sq += d * d; }
#pragma unroll
  for (int off = 16; off; off >>= 1) sq += __shfl_xor(sq, off, 32);
  float rinv = rsqrtf(sq * (1.f / CC) + LNEPS);
  __bf16* o = kv + ((size_t)b * NKVTOT + N2 + m) * CC;
#pragma unroll
  for (int i = 0; i < 8; ++i) {
    int c = lane * 8 + i;
    o[c] = (__bf16)((v[i] - mean) * rinv * g[c] + bb[c]);
  }
}

// ---------------------------------------------------------------- LN2 (unfold k=2, 1024)
__global__ void __launch_bounds__(256) ln2_kernel(const float* __restrict__ x,
                                                  const float* __restrict__ g,
                                                  const float* __restrict__ bb,
                                                  __bf16* __restrict__ a2) {
  int lane = threadIdx.x & 31, wv = threadIdx.x >> 5;
  long row = (long)blockIdx.x * 8 + wv;            // 0 .. B*N2-1
  int b = (int)(row >> 10), L = (int)(row & (N2 - 1));
  int y2 = L >> 5, x2 = L & 31;
  float v[32], s = 0.f;
#pragma unroll
  for (int i = 0; i < 32; ++i) {
    int ch = lane + 32 * i;
    int c = ch >> 2, ki = (ch >> 1) & 1, kj = ch & 1;
    int pix = (y2 * 2 + ki) * 64 + x2 * 2 + kj;
    v[i] = x[((size_t)b * NN + pix) * CC + c];
    s += v[i];
  }
#pragma unroll
  for (int off = 16; off; off >>= 1) s += __shfl_xor(s, off, 32);
  float mean = s * (1.f / 1024.f), sq = 0.f;
#pragma unroll
  for (int i = 0; i < 32; ++i) { float d = v[i] - mean; sq += d * d; }
#pragma unroll
  for (int off = 16; off; off >>= 1) sq += __shfl_xor(sq, off, 32);
  float rinv = rsqrtf(sq * (1.f / 1024.f) + LNEPS);
  __bf16* o = a2 + (size_t)row * 1024;
#pragma unroll
  for (int i = 0; i < 32; ++i) {
    int ch = lane + 32 * i;
    o[ch] = (__bf16)((v[i] - mean) * rinv * g[ch] + bb[ch]);
  }
}

// ---------------------------------------------------------------- LN3 (unfold k=4, 4096)
__global__ void __launch_bounds__(256) ln3_kernel(const float* __restrict__ x,
                                                  const float* __restrict__ g,
                                                  const float* __restrict__ bb,
                                                  __bf16* __restrict__ a3) {
  __shared__ float red[8];
  int t = threadIdx.x, lane = t & 31, wv = t >> 5;
  int row = blockIdx.x;                            // 0 .. B*N3-1
  int b = row >> 8, L = row & 255;
  int y4 = L >> 4, x4 = L & 15;
  int ki = (t >> 2) & 3, kj = t & 3;
  int pix = (y4 * 4 + ki) * 64 + x4 * 4 + kj;
  const float* xb = x + ((size_t)b * NN + pix) * CC;
  float v[16], s = 0.f;
#pragma unroll
  for (int i = 0; i < 16; ++i) { v[i] = xb[(t >> 4) + 16 * i]; s += v[i]; }
#pragma unroll
  for (int off = 16; off; off >>= 1) s += __shfl_xor(s, off, 32);
  if (lane == 0) red[wv] = s;
  __syncthreads();
  float tot = 0.f;
#pragma unroll
  for (int i = 0; i < 8; ++i) tot += red[i];
  float mean = tot * (1.f / 4096.f), sq = 0.f;
#pragma unroll
  for (int i = 0; i < 16; ++i) { float d = v[i] - mean; sq += d * d; }
#pragma unroll
  for (int off = 16; off; off >>= 1) sq += __shfl_xor(sq, off, 32);
  __syncthreads();
  if (lane == 0) red[wv] = sq;
  __syncthreads();
  float tsq = 0.f;
#pragma unroll
  for (int i = 0; i < 8; ++i) tsq += red[i];
  float rinv = rsqrtf(tsq * (1.f / 4096.f) + LNEPS);
  __bf16* o = a3 + (size_t)row * 4096;
#pragma unroll
  for (int i = 0; i < 16; ++i) {
    int ch = t + 256 * i;
    o[ch] = (__bf16)((v[i] - mean) * rinv * g[ch] + bb[ch]);
  }
}

// ---------------------------------------------------------------- GEMMs
__global__ void __launch_bounds__(128) gemm_split2(const __bf16* __restrict__ A,
                                                   const __bf16* __restrict__ Bt,
                                                   const float* __restrict__ bias,
                                                   __bf16* __restrict__ kv) {
  const int K = 1024;
  int lane = threadIdx.x & 31, wv = threadIdx.x >> 5;
  int hlf = lane >> 4, col = lane & 15;
  int tileM = (blockIdx.x * 4 + wv) * 16, tileN = blockIdx.y * 64;
  v8f acc[4]; zero4(acc);
  wmma_kloop<K>(A + (size_t)(tileM + col) * K, Bt + (size_t)(tileN + col) * K, hlf, acc);
#pragma unroll
  for (int nt = 0; nt < 4; ++nt) {
    int c = tileN + nt * 16 + col;
    float bv = bias[c];
#pragma unroll
    for (int r = 0; r < 8; ++r) {
      int m = tileM + r + 8 * hlf;
      int b = m >> 10, mm = m & 1023;
      kv[((size_t)b * NKVTOT + mm) * CC + c] = (__bf16)(acc[nt][r] + bv);
    }
  }
}

__global__ void __launch_bounds__(128) gemm_split3(const __bf16* __restrict__ A,
                                                   const __bf16* __restrict__ Bt,
                                                   const float* __restrict__ bias,
                                                   __bf16* __restrict__ s3) {
  const int K = 4096;
  int lane = threadIdx.x & 31, wv = threadIdx.x >> 5;
  int hlf = lane >> 4, col = lane & 15;
  int tileM = (blockIdx.x * 4 + wv) * 16, tileN = blockIdx.y * 64;
  v8f acc[4]; zero4(acc);
  wmma_kloop<K>(A + (size_t)(tileM + col) * K, Bt + (size_t)(tileN + col) * K, hlf, acc);
#pragma unroll
  for (int nt = 0; nt < 4; ++nt) {
    int c = tileN + nt * 16 + col;
    float bv = bias[c];
#pragma unroll
    for (int r = 0; r < 8; ++r) {
      int m = tileM + r + 8 * hlf;
      s3[(size_t)m * CC + c] = (__bf16)(acc[nt][r] + bv);
    }
  }
}

__global__ void __launch_bounds__(128) gemm_q(const __bf16* __restrict__ A,
                                              const __bf16* __restrict__ Bt,
                                              float* __restrict__ qlin,
                                              __bf16* __restrict__ qb) {
  const int K = 256;
  int lane = threadIdx.x & 31, wv = threadIdx.x >> 5;
  int hlf = lane >> 4, col = lane & 15;
  int tileM = (blockIdx.x * 4 + wv) * 16, tileN = blockIdx.y * 64;
  v8f acc[4]; zero4(acc);
  wmma_kloop<K>(A + (size_t)(tileM + col) * K, Bt + (size_t)(tileN + col) * K, hlf, acc);
#pragma unroll
  for (int nt = 0; nt < 4; ++nt) {
    int c = tileN + nt * 16 + col;
    int h = c >> 6, d = c & 63;
#pragma unroll
    for (int r = 0; r < 8; ++r) {
      int m = tileM + r + 8 * hlf;
      int b = m >> 8, mm = m & 255;
      float v = acc[nt][r];
      qlin[(size_t)m * INDIM + c] = v;
      qb[(((size_t)b * HEADS + h) * N3 + mm) * HDIM + d] = (__bf16)(v * 0.125f);
    }
  }
}

__global__ void __launch_bounds__(128) gemm_kv(const __bf16* __restrict__ A,
                                               const __bf16* __restrict__ Bt,
                                               __bf16* __restrict__ kb,
                                               __bf16* __restrict__ vt) {
  const int K = 256;
  int lane = threadIdx.x & 31, wv = threadIdx.x >> 5;
  int hlf = lane >> 4, col = lane & 15;
  int tileM = (blockIdx.x * 4 + wv) * 16, tileN = blockIdx.y * 64;
  v8f acc[4]; zero4(acc);
  wmma_kloop<K>(A + (size_t)(tileM + col) * K, Bt + (size_t)(tileN + col) * K, hlf, acc);
#pragma unroll
  for (int nt = 0; nt < 4; ++nt) {
    int c = tileN + nt * 16 + col;
#pragma unroll
    for (int r = 0; r < 8; ++r) {
      int m = tileM + r + 8 * hlf;
      int b = m / NKVTOT, mm = m % NKVTOT;
      __bf16 bv = (__bf16)acc[nt][r];
      if (c < INDIM) {
        int h = c >> 6, d = c & 63;
        kb[(((size_t)b * HEADS + h) * NKVTOT + mm) * HDIM + d] = bv;
      } else {
        int c2 = c - INDIM, h = c2 >> 6, d = c2 & 63;
        vt[(((size_t)b * HEADS + h) * HDIM + d) * (size_t)NKVTOT + mm] = bv;
      }
    }
  }
}

__global__ void __launch_bounds__(128) gemm_proj(const __bf16* __restrict__ A,
                                                 const __bf16* __restrict__ Bt,
                                                 const float* __restrict__ bias,
                                                 const float* __restrict__ qlin,
                                                 float* __restrict__ ar) {
  const int K = 512;
  int lane = threadIdx.x & 31, wv = threadIdx.x >> 5;
  int hlf = lane >> 4, col = lane & 15;
  int tileM = (blockIdx.x * 4 + wv) * 16, tileN = blockIdx.y * 64;
  v8f acc[4]; zero4(acc);
  wmma_kloop<K>(A + (size_t)(tileM + col) * K, Bt + (size_t)(tileN + col) * K, hlf, acc);
#pragma unroll
  for (int nt = 0; nt < 4; ++nt) {
    int c = tileN + nt * 16 + col;
    float bv = bias[c];
#pragma unroll
    for (int r = 0; r < 8; ++r) {
      int m = tileM + r + 8 * hlf;
      ar[(size_t)m * INDIM + c] = acc[nt][r] + bv + qlin[(size_t)m * INDIM + c];
    }
  }
}

// ---------------------------------------------------------------- flash attention
// grid (B*HEADS, N3/64); 4 waves; each wave owns 16 query rows x 64 d-dims.
// K/V chunks double-buffered in LDS, filled with async global->LDS copies.
__global__ void __launch_bounds__(128) attn_kernel(const __bf16* __restrict__ qb,
                                                   const __bf16* __restrict__ kb,
                                                   const __bf16* __restrict__ vtb,
                                                   __bf16* __restrict__ ob) {
  __shared__ __bf16 lK[2][64 * 64];   // [buf][kv_chunk][d]
  __shared__ __bf16 lVT[2][64 * 64];  // [buf][d][kv_chunk]
  __shared__ __bf16 lP[4][16 * 64];   // per-wave probs [qrow][kv_chunk]
  int tid = threadIdx.x, lane = tid & 31, wv = tid >> 5;
  int hlf = lane >> 4, col = lane & 15;
  int bh = blockIdx.x, mblk = blockIdx.y;

  const __bf16* Q  = qb  + ((size_t)bh * N3 + mblk * 64 + wv * 16 + col) * HDIM;
  const __bf16* Kp = kb  + (size_t)bh * NKVTOT * HDIM;
  const __bf16* Vt = vtb + (size_t)bh * HDIM * (size_t)NKVTOT;

  auto load_chunk = [&](int kc, int buf) {
#pragma unroll
    for (int i = 0; i < 4; ++i) {
      int q = tid * 4 + i, r = q >> 3, seg = q & 7;
#if USE_ASYNC_LDS
      unsigned lk = (unsigned)(size_t)(const void*)&lK[buf][r * 64 + seg * 8];
      unsigned gk = (unsigned)((kc + r) * (HDIM * 2) + seg * 16);
      async_ld_b128(lk, gk, Kp);
      unsigned lv = (unsigned)(size_t)(const void*)&lVT[buf][r * 64 + seg * 8];
      unsigned gv = (unsigned)(r * (NKVTOT * 2) + (kc + seg * 8) * 2);
      async_ld_b128(lv, gv, Vt);
#else
      *(v8bf*)(&lK[buf][r * 64 + seg * 8]) =
          *(const v8bf*)(Kp + (size_t)(kc + r) * HDIM + seg * 8);
      *(v8bf*)(&lVT[buf][r * 64 + seg * 8]) =
          *(const v8bf*)(Vt + (size_t)r * NKVTOT + kc + seg * 8);
#endif
    }
  };

  v16bf qf[2];
#pragma unroll
  for (int ks = 0; ks < 2; ++ks)
    qf[ks] = frag_ab(Q + ks * 32 + 16 * hlf, Q + ks * 32 + 8 + 16 * hlf);

  v8f o[4]; zero4(o);
  float mr[8], lr[8];
#pragma unroll
  for (int r = 0; r < 8; ++r) { mr[r] = -1e30f; lr[r] = 0.f; }

  load_chunk(0, 0);
#if USE_ASYNC_LDS
  wait_async0();
#endif
  __syncthreads();

  int cur = 0;
  for (int kc = 0; kc < NKVTOT; kc += 64) {
    if (kc + 64 < NKVTOT) load_chunk(kc + 64, cur ^ 1);  // prefetch next chunk

    v8f s[4]; zero4(s);
#pragma unroll
    for (int nt = 0; nt < 4; ++nt) {
      const __bf16* bp = &lK[cur][(nt * 16 + col) * 64];
#pragma unroll
      for (int ks = 0; ks < 2; ++ks)
        s[nt] = __builtin_amdgcn_wmma_f32_16x16x32_bf16(
            false, qf[ks], false, frag_c16(bp + ks * 32 + 16 * hlf),
            (short)0, s[nt], false, false);
    }

    __bf16* Pw = lP[wv];
#pragma unroll
    for (int r = 0; r < 8; ++r) {
      float x0 = fmaxf(fmaxf(s[0][r], s[1][r]), fmaxf(s[2][r], s[3][r]));
#pragma unroll
      for (int off = 8; off; off >>= 1) x0 = fmaxf(x0, __shfl_xor(x0, off, 32));
      float mnew = fmaxf(mr[r], x0);
      float alpha = __expf(mr[r] - mnew);
      mr[r] = mnew;
      float rs = 0.f;
      int rowoff = (r + 8 * hlf) * 64;
#pragma unroll
      for (int nt = 0; nt < 4; ++nt) {
        float p = __expf(s[nt][r] - mnew);
        rs += p;
        Pw[rowoff + nt * 16 + col] = (__bf16)p;
        o[nt][r] *= alpha;
      }
#pragma unroll
      for (int off = 8; off; off >>= 1) rs += __shfl_xor(rs, off, 32);
      lr[r] = lr[r] * alpha + rs;
    }
    __syncthreads();   // lP visible (and keeps waves in step)

    const __bf16* Pr = Pw + col * 64;
#pragma unroll
    for (int ks = 0; ks < 2; ++ks) {
      v16bf pf = frag_ab(Pr + ks * 32 + 16 * hlf, Pr + ks * 32 + 8 + 16 * hlf);
#pragma unroll
      for (int nt = 0; nt < 4; ++nt)
        o[nt] = __builtin_amdgcn_wmma_f32_16x16x32_bf16(
            false, pf, false,
            frag_c16(&lVT[cur][(nt * 16 + col) * 64 + ks * 32 + 16 * hlf]),
            (short)0, o[nt], false, false);
    }

#if USE_ASYNC_LDS
    wait_async0();     // my prefetch landed
#endif
    __syncthreads();   // everyone done with buf[cur]; next iter may overwrite it
    cur ^= 1;
  }

  int b = bh >> 3, h = bh & 7;
#pragma unroll
  for (int nt = 0; nt < 4; ++nt) {
    int d = h * 64 + nt * 16 + col;
#pragma unroll
    for (int r = 0; r < 8; ++r) {
      int qrow = mblk * 64 + wv * 16 + r + 8 * hlf;
      ob[((size_t)b * N3 + qrow) * INDIM + d] = (__bf16)(o[nt][r] / lr[r]);
    }
  }
}

// ---------------------------------------------------------------- fold + LN(32) + MLP
__global__ void __launch_bounds__(256) fold_mlp(const float* __restrict__ ar,
                                                const float* __restrict__ g,
                                                const float* __restrict__ bb,
                                                const float* __restrict__ w1,
                                                const float* __restrict__ b1,
                                                const float* __restrict__ w2,
                                                const float* __restrict__ b2,
                                                float* __restrict__ out) {
  __shared__ float sw1[1024], sw2[1024], sb1[32], sb2[32], sg[32], sbv[32];
  int t = threadIdx.x;
  for (int i = t; i < 1024; i += 256) { sw1[i] = w1[i]; sw2[i] = w2[i]; }
  if (t < 32) { sb1[t] = b1[t]; sb2[t] = b2[t]; sg[t] = g[t]; sbv[t] = bb[t]; }
  __syncthreads();
  long row = (long)blockIdx.x * 256 + t;           // 0 .. B*NN-1
  int b = (int)(row >> 12), pix = (int)(row & 4095);
  int y = pix >> 6, xx = pix & 63;
  int y4 = y >> 2, ki = y & 3, x4 = xx >> 2, kj = xx & 3;
  int L = y4 * 16 + x4;
  const float* src = ar + ((size_t)b * N3 + L) * INDIM + ki * 4 + kj;
  float fr[32], s = 0.f;
#pragma unroll
  for (int cf = 0; cf < 32; ++cf) { fr[cf] = src[cf * 16]; s += fr[cf]; }
  float mean = s * (1.f / 32.f), sq = 0.f;
#pragma unroll
  for (int cf = 0; cf < 32; ++cf) { float d = fr[cf] - mean; sq += d * d; }
  float rinv = rsqrtf(sq * (1.f / 32.f) + LNEPS);
  float ln[32], h1[32];
#pragma unroll
  for (int cf = 0; cf < 32; ++cf) ln[cf] = (fr[cf] - mean) * rinv * sg[cf] + sbv[cf];
#pragma unroll
  for (int j = 0; j < 32; ++j) {
    float a = sb1[j];
#pragma unroll
    for (int i = 0; i < 32; ++i) a += ln[i] * sw1[i * 32 + j];
    h1[j] = fmaxf(a, 0.f);
  }
  float* orow = out + (size_t)row * 32;
#pragma unroll
  for (int j = 0; j < 32; ++j) {
    float a = sb2[j];
#pragma unroll
    for (int i = 0; i < 32; ++i) a += h1[i] * sw2[i * 32 + j];
    orow[j] = fr[j] + a;
  }
}

// ---------------------------------------------------------------- launcher
extern "C" void kernel_launch(void* const* d_in, const int* in_sizes, int n_in,
                              void* d_out, int out_size, void* d_ws, size_t ws_size,
                              hipStream_t stream) {
  (void)in_sizes; (void)n_in; (void)out_size; (void)ws_size;
  const float* x      = (const float*)d_in[0];
  const float* ln1_g  = (const float*)d_in[1];
  const float* ln1_b  = (const float*)d_in[2];
  const float* ln2_g  = (const float*)d_in[3];
  const float* ln2_b  = (const float*)d_in[4];
  const float* ln3_g  = (const float*)d_in[5];
  const float* ln3_b  = (const float*)d_in[6];
  const float* fc2_w  = (const float*)d_in[7];
  const float* fc2_b  = (const float*)d_in[8];
  const float* fc3_w  = (const float*)d_in[9];
  const float* fc3_b  = (const float*)d_in[10];
  const float* q_w    = (const float*)d_in[11];
  const float* kv_w   = (const float*)d_in[12];
  const float* proj_w = (const float*)d_in[13];
  const float* proj_b = (const float*)d_in[14];
  const float* n2_g   = (const float*)d_in[15];
  const float* n2_b   = (const float*)d_in[16];
  const float* mlp1_w = (const float*)d_in[17];
  const float* mlp1_b = (const float*)d_in[18];
  const float* mlp2_w = (const float*)d_in[19];
  const float* mlp2_b = (const float*)d_in[20];

  char* ws = (char*)d_ws;
  size_t off = 0;
  __bf16* fc2_wT  = (__bf16*)(ws + off); off += (size_t)1024 * 256 * 2;
  __bf16* fc3_wT  = (__bf16*)(ws + off); off += (size_t)4096 * 256 * 2;
  __bf16* q_wT    = (__bf16*)(ws + off); off += (size_t)256 * 512 * 2;
  __bf16* kv_wT   = (__bf16*)(ws + off); off += (size_t)256 * 1024 * 2;
  __bf16* proj_wT = (__bf16*)(ws + off); off += (size_t)512 * 512 * 2;
  __bf16* kv_bf   = (__bf16*)(ws + off); off += (size_t)BATCH * NKVTOT * CC * 2;
  __bf16* a2      = (__bf16*)(ws + off); off += (size_t)BATCH * N2 * 1024 * 2;
  __bf16* a3      = (__bf16*)(ws + off); off += (size_t)BATCH * N3 * 4096 * 2;
  __bf16* s3      = (__bf16*)(ws + off); off += (size_t)BATCH * N3 * CC * 2;
  float*  qlin    = (float*) (ws + off); off += (size_t)BATCH * N3 * INDIM * 4;
  __bf16* qb      = (__bf16*)(ws + off); off += (size_t)BATCH * N3 * INDIM * 2;
  __bf16* kb      = (__bf16*)(ws + off); off += (size_t)BATCH * HEADS * NKVTOT * HDIM * 2;
  __bf16* vt      = (__bf16*)(ws + off); off += (size_t)BATCH * HEADS * NKVTOT * HDIM * 2;
  __bf16* ob      = (__bf16*)(ws + off); off += (size_t)BATCH * N3 * INDIM * 2;
  float*  ares    = (float*) (ws + off); off += (size_t)BATCH * N3 * INDIM * 4;

  // weights -> transposed bf16
  wcvt_kernel<<<(1024 * 256 + 255) / 256, 256, 0, stream>>>(fc2_w, fc2_wT, 1024, 256);
  wcvt_kernel<<<(4096 * 256 + 255) / 256, 256, 0, stream>>>(fc3_w, fc3_wT, 4096, 256);
  wcvt_kernel<<<(256 * 512 + 255) / 256, 256, 0, stream>>>(q_w, q_wT, 256, 512);
  wcvt_kernel<<<(256 * 1024 + 255) / 256, 256, 0, stream>>>(kv_w, kv_wT, 256, 1024);
  wcvt_kernel<<<(512 * 512 + 255) / 256, 256, 0, stream>>>(proj_w, proj_wT, 512, 512);

  // layernorm / unfold stages
  ln1_kernel<<<BATCH * NN / 8, 256, 0, stream>>>(x, ln1_g, ln1_b, kv_bf);
  ln2_kernel<<<BATCH * N2 / 8, 256, 0, stream>>>(x, ln2_g, ln2_b, a2);
  ln3_kernel<<<BATCH * N3, 256, 0, stream>>>(x, ln3_g, ln3_b, a3);

  // GEMMs (WMMA bf16, fp32 accumulate)
  gemm_split2<<<dim3(BATCH * N2 / 64, 256 / 64), 128, 0, stream>>>(a2, fc2_wT, fc2_b, kv_bf);
  gemm_split3<<<dim3(BATCH * N3 / 64, 256 / 64), 128, 0, stream>>>(a3, fc3_wT, fc3_b, s3);
  gemm_q<<<dim3(BATCH * N3 / 64, INDIM / 64), 128, 0, stream>>>(s3, q_wT, qlin, qb);
  gemm_kv<<<dim3(BATCH * NKVTOT / 64, 1024 / 64), 128, 0, stream>>>(kv_bf, kv_wT, kb, vt);

  // flash attention (async LDS double-buffered)
  attn_kernel<<<dim3(BATCH * HEADS, N3 / 64), 128, 0, stream>>>(qb, kb, vt, ob);

  // output projection + residual
  gemm_proj<<<dim3(BATCH * N3 / 64, INDIM / 64), 128, 0, stream>>>(ob, proj_wT, proj_b, qlin, ares);

  // fold + LN(32) + MLP + residual -> d_out
  fold_mlp<<<BATCH * NN / 256, 256, 0, stream>>>(ares, n2_g, n2_b, mlp1_w, mlp1_b,
                                                 mlp2_w, mlp2_b, (float*)d_out);
}
